// FullCPNN_51539607553070
// MI455X (gfx1250) — matrix operand order repaired and checked by reading, not compiled, not run on hardware
//
#include <hip/hip_runtime.h>
#include <hip/hip_bf16.h>

// Problem constants (from reference): B=8192, D=512, H=16384, O=1000
#define PB 8192
#define PD 512
#define PH 16384
#define PO 1000

// Tiling for the GEMM+argmin kernel
#define MB 128      // M rows per block (8 waves x 16 rows)
#define NT 128      // N columns per inner tile (8 WMMA subtiles of 16)
#define HS 2048     // H strip per block (grid.y = PH/HS)
#define KT 32       // K per step (bf16 WMMA depth)
#define RS 40       // padded LDS row stride in bf16 elements (80B, 16B-aligned, conflict-free)

typedef __attribute__((ext_vector_type(16))) __bf16   v16bf;
typedef __attribute__((ext_vector_type(8)))  float    v8f;
typedef __attribute__((ext_vector_type(4)))  unsigned u32x4;
typedef __attribute__((ext_vector_type(2)))  unsigned u32x2;
typedef __attribute__((ext_vector_type(4)))  float    f32x4;
typedef __attribute__((ext_vector_type(4)))  int      i32x4;
// Address-space qualified views for the async global->LDS builtin
typedef __attribute__((address_space(1))) i32x4 i32x4_g;   // global (prints as __device__)
typedef __attribute__((address_space(3))) i32x4 i32x4_l;   // LDS

union FragBF {
    v16bf v;
    u32x4 q[2];
};

__device__ __forceinline__ unsigned short f2bf_rn(float f) {
    unsigned u = __float_as_uint(f);
    unsigned r = u + 0x7FFFu + ((u >> 16) & 1u);   // round-to-nearest-even
    return (unsigned short)(r >> 16);
}
__device__ __forceinline__ float bf2f(unsigned short h) {
    return __uint_as_float(((unsigned)h) << 16);
}
// monotone float -> uint mapping (min over uint == min over float)
__device__ __forceinline__ unsigned orderF(float f) {
    unsigned b = __float_as_uint(f);
    return (b & 0x80000000u) ? ~b : (b | 0x80000000u);
}

// ---- CDNA5 async global->LDS copy (ASYNCcnt path), with safe fallback -----
#if __has_builtin(__builtin_amdgcn_global_load_async_to_lds_b128)
#define HAVE_ASYNC_LDS 1
#define ASYNC_CP_B128(gsrc, ldst)                                             \
    __builtin_amdgcn_global_load_async_to_lds_b128(                           \
        (i32x4_g*)(gsrc), (i32x4_l*)(ldst), 0, 0)
__device__ __forceinline__ void async_wait0() {
#if __has_builtin(__builtin_amdgcn_s_wait_asynccnt)
    __builtin_amdgcn_s_wait_asynccnt(0);
#else
    asm volatile("s_wait_asynccnt 0" ::: "memory");
#endif
}
#else
#define HAVE_ASYNC_LDS 0
#endif

// ---------------------------------------------------------------- kernel 0
__global__ void cpnn_init_keys(unsigned long long* __restrict__ keys) {
    int i = blockIdx.x * blockDim.x + threadIdx.x;
    if (i < PB) keys[i] = ~0ULL;
}

// ---------------------------------------------------------------- kernel 1
// One-pass f32 -> bf16 (hi, lo) split: v = hi + lo to ~2^-16 relative error.
// Removes all conversion VALU from the GEMM hot loop and halves its traffic.
__global__ __launch_bounds__(256) void cpnn_split(
    const float* __restrict__ src, unsigned short* __restrict__ hi,
    unsigned short* __restrict__ lo, int n4)
{
    int i = blockIdx.x * blockDim.x + threadIdx.x;
    if (i >= n4) return;
    f32x4 v = ((const f32x4*)src)[i];
    unsigned short h0 = f2bf_rn(v.x), h1 = f2bf_rn(v.y);
    unsigned short h2 = f2bf_rn(v.z), h3 = f2bf_rn(v.w);
    unsigned short l0 = f2bf_rn(v.x - bf2f(h0));
    unsigned short l1 = f2bf_rn(v.y - bf2f(h1));
    unsigned short l2 = f2bf_rn(v.z - bf2f(h2));
    unsigned short l3 = f2bf_rn(v.w - bf2f(h3));
    u32x2 ph = { (unsigned)h0 | ((unsigned)h1 << 16),
                 (unsigned)h2 | ((unsigned)h3 << 16) };
    u32x2 pl = { (unsigned)l0 | ((unsigned)l1 << 16),
                 (unsigned)l2 | ((unsigned)l3 << 16) };
    ((u32x2*)hi)[i] = ph;
    ((u32x2*)lo)[i] = pl;
}

// ---------------------------------------------------------------- kernel 2
// w2[h] = sum_k W[h,k]^2 ; one wave32 per row.
__global__ __launch_bounds__(256) void cpnn_w2(const float* __restrict__ W,
                                               float* __restrict__ w2) {
    int lane = threadIdx.x & 31;
    int wave = threadIdx.x >> 5;
    int row  = blockIdx.x * 8 + wave;
    const float* p = W + (size_t)row * PD;
    float s = 0.f;
    #pragma unroll
    for (int i = 0; i < PD / 32; ++i) {
        float v = p[lane + 32 * i];
        s += v * v;
    }
    #pragma unroll
    for (int m = 1; m < 32; m <<= 1) s += __shfl_xor(s, m, 32);
    if (lane == 0) w2[row] = s;
}

// ---------------------------------------------------------------- kernel 3
// Tiled bf16x3 WMMA GEMM (x @ W.T) fused with running argmin of
// score[h] = w2[h] - 2*dot(x_b, w_h). Winners merged via u64 atomicMin.
// Inputs are pre-split bf16 planes; staging is a pure b128 copy (async if
// the toolchain exposes global_load_async_to_lds).
__global__ __launch_bounds__(256) void cpnn_gemm_argmin(
    const unsigned short* __restrict__ xh, const unsigned short* __restrict__ xl,
    const unsigned short* __restrict__ wh, const unsigned short* __restrict__ wl,
    const float* __restrict__ w2, unsigned long long* __restrict__ keys)
{
    __shared__ __align__(16) unsigned short sXh[MB * RS];
    __shared__ __align__(16) unsigned short sXl[MB * RS];
    __shared__ __align__(16) unsigned short sWh[NT * RS];
    __shared__ __align__(16) unsigned short sWl[NT * RS];

    const int tid   = threadIdx.x;
    const int lane  = tid & 31;
    const int wave  = tid >> 5;          // 8 waves
    const int ln    = lane & 15;
    const int hl    = lane >> 4;         // half-wave select
    const int mBase = blockIdx.x * MB;
    const int hBase = blockIdx.y * HS;

    // staging role: each thread copies one 16B chunk (8 bf16) per plane/pass
    const int srow = tid >> 2;           // 0..63
    const int sc8  = (tid & 3) * 8;      // 0,8,16,24

    unsigned long long best[8];
    #pragma unroll
    for (int r = 0; r < 8; ++r) best[r] = ~0ULL;

    const v8f zacc = {};

    for (int nt = 0; nt < HS; nt += NT) {
        v8f acc[8];
        #pragma unroll
        for (int n = 0; n < 8; ++n) acc[n] = zacc;

        for (int k0 = 0; k0 < PD; k0 += KT) {
            // ---- stage 128x32 bf16 tiles of x(hi/lo) and W(hi/lo) ----
            #pragma unroll
            for (int p = 0; p < 2; ++p) {
                int row = srow + p * 64;
                size_t gx = (size_t)(mBase + row) * PD + k0 + sc8;
                size_t gw = (size_t)(hBase + nt + row) * PD + k0 + sc8;
                int li = row * RS + sc8;
#if HAVE_ASYNC_LDS
                ASYNC_CP_B128(xh + gx, &sXh[li]);
                ASYNC_CP_B128(xl + gx, &sXl[li]);
                ASYNC_CP_B128(wh + gw, &sWh[li]);
                ASYNC_CP_B128(wl + gw, &sWl[li]);
#else
                *(u32x4*)&sXh[li] = *(const u32x4*)(xh + gx);
                *(u32x4*)&sXl[li] = *(const u32x4*)(xl + gx);
                *(u32x4*)&sWh[li] = *(const u32x4*)(wh + gw);
                *(u32x4*)&sWl[li] = *(const u32x4*)(wl + gw);
                if (k0 + KT < PD) {   // prefetch next K tile
                    __builtin_prefetch(xh + gx + KT, 0, 3);
                    __builtin_prefetch(wh + gw + KT, 0, 3);
                }
#endif
            }
#if HAVE_ASYNC_LDS
            async_wait0();
#endif
            __syncthreads();

            // A fragments (16x32 bf16, ISA layout: lane ln holds row, kb=8*hl,
            // K chunks [kb,kb+8) and [kb+16,kb+24))
            FragBF Ah, Al;
            {
                int arow = wave * 16 + ln;
                int kb   = hl * 8;
                Ah.q[0] = *(const u32x4*)&sXh[arow * RS + kb];
                Ah.q[1] = *(const u32x4*)&sXh[arow * RS + kb + 16];
                Al.q[0] = *(const u32x4*)&sXl[arow * RS + kb];
                Al.q[1] = *(const u32x4*)&sXl[arow * RS + kb + 16];
            }
            #pragma unroll
            for (int n = 0; n < 8; ++n) {
                // B fragment (32x16): same K layout as A (kb=8*hl, chunks [kb,kb+8) and [kb+16,kb+24)),
                // contiguous along W's row (W.T column) -> two b128 LDS loads.
                FragBF Bh, Bl;
                int brow = n * 16 + ln;
                int kc   = hl * 8;
                Bh.q[0] = *(const u32x4*)&sWh[brow * RS + kc];
                Bh.q[1] = *(const u32x4*)&sWh[brow * RS + kc + 16];
                Bl.q[0] = *(const u32x4*)&sWl[brow * RS + kc];
                Bl.q[1] = *(const u32x4*)&sWl[brow * RS + kc + 16];
                // bf16x3 split: hi*hi + hi*lo + lo*hi, f32 accumulate
                acc[n] = __builtin_amdgcn_wmma_f32_16x16x32_bf16(
                    false, Ah.v, false, Bh.v, (short)0, acc[n], false, false);
                acc[n] = __builtin_amdgcn_wmma_f32_16x16x32_bf16(
                    false, Ah.v, false, Bl.v, (short)0, acc[n], false, false);
                acc[n] = __builtin_amdgcn_wmma_f32_16x16x32_bf16(
                    false, Al.v, false, Bh.v, (short)0, acc[n], false, false);
            }
            __syncthreads();  // LDS consumed; next iteration may overwrite
        }

        // fold this N-tile into per-lane running argmin.
        // C layout: VGPR r, lanes 0-15 -> M = r, lanes 16-31 -> M = r+8; N = ln.
        #pragma unroll
        for (int n = 0; n < 8; ++n) {
            int col  = hBase + nt + n * 16 + ln;
            float wv = w2[col];
            #pragma unroll
            for (int r = 0; r < 8; ++r) {
                float s = wv - 2.0f * acc[n][r];
                unsigned long long key =
                    ((unsigned long long)orderF(s) << 32) | (unsigned)col;
                if (key < best[r]) best[r] = key;
            }
        }
    }

    // reduce across the 16 lanes sharing each row, then global merge
    #pragma unroll
    for (int r = 0; r < 8; ++r) {
        unsigned long long k = best[r];
        #pragma unroll
        for (int m = 1; m < 16; m <<= 1) {
            unsigned lo = __shfl_xor((unsigned)(k & 0xffffffffu), m, 32);
            unsigned hi = __shfl_xor((unsigned)(k >> 32), m, 32);
            unsigned long long o = ((unsigned long long)hi << 32) | lo;
            if (o < k) k = o;
        }
        if (ln == 0) {
            int row = mBase + wave * 16 + r + 8 * hl;
            atomicMin(&keys[row], k);
        }
    }
}

// ---------------------------------------------------------------- kernel 4
// output[b,:] = G_fwd[:,w]; recos[b,:] = G_rev[:,w]; winners[b] = w (as float)
__global__ __launch_bounds__(256) void cpnn_gather(
    const unsigned long long* __restrict__ keys,
    const float* __restrict__ G_fwd, const float* __restrict__ G_rev,
    float* __restrict__ out)
{
    int b = blockIdx.x;
    unsigned w = (unsigned)(keys[b] & 0xffffffffu);
    float* outp = out + (size_t)b * PO;
    float* recp = out + (size_t)PB * PO + (size_t)b * PD;
    for (int o = threadIdx.x; o < PO; o += 256)
        outp[o] = G_fwd[(size_t)o * PH + w];
    for (int d = threadIdx.x; d < PD; d += 256)
        recp[d] = G_rev[(size_t)d * PH + w];
    if (threadIdx.x == 0)
        out[(size_t)PB * PO + (size_t)PB * PD + b] = (float)w;
}

// ----------------------------------------------------------------
extern "C" void kernel_launch(void* const* d_in, const int* in_sizes, int n_in,
                              void* d_out, int out_size, void* d_ws, size_t ws_size,
                              hipStream_t stream) {
    const float* x     = (const float*)d_in[0];  // (B, D)
    const float* W     = (const float*)d_in[1];  // (H, D)
    const float* G_fwd = (const float*)d_in[2];  // (O, H)
    const float* G_rev = (const float*)d_in[3];  // (D, H)
    float* out = (float*)d_out;

    // ws layout (bytes):
    //   [0,              PH*4)        w2
    //   [PH*4,           +PB*8)       packed (score|idx) keys
    //   then bf16 planes: xh, xl (PB*PD*2 each), wh, wl (PH*PD*2 each)
    char* base = (char*)d_ws;
    float* w2 = (float*)base;
    unsigned long long* keys = (unsigned long long*)(base + (size_t)PH * 4);
    unsigned short* xh = (unsigned short*)(base + (size_t)PH * 4 + (size_t)PB * 8);
    unsigned short* xl = xh + (size_t)PB * PD;
    unsigned short* wh = xl + (size_t)PB * PD;
    unsigned short* wl = wh + (size_t)PH * PD;

    cpnn_init_keys<<<(PB + 255) / 256, 256, 0, stream>>>(keys);
    cpnn_split<<<(PB * PD / 4 + 255) / 256, 256, 0, stream>>>(x, xh, xl, PB * PD / 4);
    cpnn_split<<<(PH * PD / 4 + 255) / 256, 256, 0, stream>>>(W, wh, wl, PH * PD / 4);
    cpnn_w2<<<PH / 8, 256, 0, stream>>>(W, w2);
    dim3 grid(PB / MB, PH / HS);
    cpnn_gemm_argmin<<<grid, 256, 0, stream>>>(xh, xl, wh, wl, w2, keys);
    cpnn_gather<<<PB, 256, 0, stream>>>(keys, G_fwd, G_rev, out);
}
